// GroupedQueryAttention_34299608825973
// MI455X (gfx1250) — compile-verified, matrix operand order checked by reading, not run on hardware
//
#include <hip/hip_runtime.h>

// ---------------------------------------------------------------------------
// Grouped-Query Attention for gfx1250 (MI455X), wave32 + v_wmma_f32_16x16x32_bf16
// All WMMA operands pre-packed so every fragment load is 2x global_load_b128.
// ---------------------------------------------------------------------------

typedef __attribute__((ext_vector_type(16))) __bf16 v16bf;
typedef __attribute__((ext_vector_type(8)))  __bf16 v8bf;
typedef __attribute__((ext_vector_type(4)))  __bf16 v4bf;
typedef __attribute__((ext_vector_type(8)))  float  v8f;
typedef __attribute__((ext_vector_type(4)))  float  v4f;

constexpr int T_SEQ = 2048;
constexpr int DM    = 1024;
constexpr int HD    = 64;
constexpr int NH    = 16;
constexpr int NKV   = 4;

__device__ __forceinline__ v8f zero8() {
  v8f z;
#pragma unroll
  for (int i = 0; i < 8; ++i) z[i] = 0.0f;
  return z;
}

__device__ __forceinline__ v8f wmma_bf16(v16bf a, v16bf b, v8f c) {
  return __builtin_amdgcn_wmma_f32_16x16x32_bf16(false, a, false, b, (short)0, c,
                                                 false, false);
}

// 16-bit fragment loader (A-operand layout; B-operand with lane = N and a
// row-contiguous [N][K] source uses the identical mapping).
// Per ISA 7.12.2 the lane's 16 half-slots are two contiguous 8-element runs:
//   k = 8*side + 0..7   and   k = 16 + 8*side + 0..7
// -> exactly two b128 loads per lane.
__device__ __forceinline__ v16bf load_frag_row(const __bf16* __restrict__ src,
                                               int ld) {
  const int lane = threadIdx.x & 31, m = lane & 15, side = lane >> 4;
  const __bf16* p = src + (size_t)m * ld + 8 * side;
  v8bf lo = *(const v8bf*)(p);
  v8bf hi = *(const v8bf*)(p + 16);
  return __builtin_shufflevector(lo, hi, 0, 1, 2, 3, 4, 5, 6, 7, 8, 9, 10, 11,
                                 12, 13, 14, 15);
}

// ---------------------------------------------------------------------------
// Pre-pack kernels (one-time): fp32 -> bf16 convert, and transpose-convert.
// ---------------------------------------------------------------------------
__global__ __launch_bounds__(256) void cvt_bf16_kernel(
    const float* __restrict__ src, __bf16* __restrict__ dst, int n4) {
  const int i = blockIdx.x * 256 + threadIdx.x;
  if (i >= n4) return;
  v4f v = *(const v4f*)(src + (size_t)i * 4);
  v4bf o;
#pragma unroll
  for (int j = 0; j < 4; ++j) o[j] = (__bf16)v[j];
  *(v4bf*)(dst + (size_t)i * 4) = o;
}

// dst[m][c][r] = src[m][r][c]  (batched RxC -> CxR transpose, fp32 -> bf16)
__global__ __launch_bounds__(256) void transpose_cvt_kernel(
    const float* __restrict__ src, __bf16* __restrict__ dst, int R, int C,
    int total) {
  const int i = blockIdx.x * 256 + threadIdx.x;
  if (i >= total) return;
  const int mc = R * C;
  const int m = i / mc;
  const int rem = i - m * mc;
  const int c = rem / R;
  const int r = rem - c * R;
  dst[i] = (__bf16)src[(size_t)m * mc + (size_t)r * C + c];
}

// ---------------------------------------------------------------------------
// Kernel 1: fused Q/K/V projection + RoPE + Q-scale.
// grid = (T/16, 6), block = 128 (4 waves); wave handles one output matrix o:
//   o in [0,16)  -> Q head o       (rope + 1/sqrt(64)), stored [t][h]
//   o in [16,20) -> K head o-16    (rope),               stored [t][h]
//   o in [20,24) -> V head o-20                          stored TRANSPOSED [h][t]
// ---------------------------------------------------------------------------
__global__ __launch_bounds__(128) void qkv_rope_kernel(
    const __bf16* __restrict__ xb, const int* __restrict__ pos,
    const __bf16* __restrict__ wqT, const __bf16* __restrict__ wkvT,
    __bf16* __restrict__ Qb, __bf16* __restrict__ Kb, __bf16* __restrict__ Vt) {
  const int o  = blockIdx.y * 4 + (threadIdx.x >> 5);
  const int m0 = blockIdx.x * 16;
  // wkvT flat (2,4,HD,DM): K heads 0..3, V heads 4..7 (each [h][d])
  const __bf16* wT = (o < NH) ? (wqT + (size_t)o * HD * DM)
                              : (wkvT + (size_t)(o - NH) * HD * DM);
  v8f acc[4];
#pragma unroll
  for (int nn = 0; nn < 4; ++nn) acc[nn] = zero8();

  for (int kk = 0; kk < DM; kk += 32) {
    v16bf a = load_frag_row(xb + (size_t)m0 * DM + kk, DM);
#pragma unroll
    for (int nn = 0; nn < 4; ++nn) {
      v16bf b = load_frag_row(wT + (size_t)(nn * 16) * DM + kk, DM);
      acc[nn] = wmma_bf16(a, b, acc[nn]);
    }
  }

  const int lane = threadIdx.x & 31, col = lane & 15, side = lane >> 4;
  if (o < NH + NKV) {  // RoPE for Q and K
#pragma unroll
    for (int r = 0; r < 8; ++r) {
      const int t = m0 + r + 8 * side;
      const float p = (float)pos[t];
#pragma unroll
      for (int j = 0; j < 2; ++j) {
        const float h = (float)(j * 16 + col);  // first-half index 0..31
        const float theta = p * __powf(10000.0f, -h * (1.0f / 32.0f));
        float sn, cs;
        __sincosf(theta, &sn, &cs);
        const float f  = acc[j][r];
        const float s2 = acc[j + 2][r];
        acc[j][r]     = f * cs - s2 * sn;
        acc[j + 2][r] = s2 * cs + f * sn;
      }
    }
  }
  if (o < NH) {  // Q: scale + store [t][h]
    __bf16* dst = Qb + (size_t)o * T_SEQ * HD;
#pragma unroll
    for (int nn = 0; nn < 4; ++nn)
#pragma unroll
      for (int r = 0; r < 8; ++r) {
        const int t = m0 + r + 8 * side;
        dst[(size_t)t * HD + nn * 16 + col] = (__bf16)(acc[nn][r] * 0.125f);
      }
  } else if (o < NH + NKV) {  // K: store [t][h]
    __bf16* dst = Kb + (size_t)(o - NH) * T_SEQ * HD;
#pragma unroll
    for (int nn = 0; nn < 4; ++nn)
#pragma unroll
      for (int r = 0; r < 8; ++r) {
        const int t = m0 + r + 8 * side;
        dst[(size_t)t * HD + nn * 16 + col] = (__bf16)acc[nn][r];
      }
  } else {  // V: store transposed [h][t] so PV B-fragments are row-contiguous
    __bf16* dst = Vt + (size_t)(o - NH - NKV) * HD * T_SEQ;
#pragma unroll
    for (int nn = 0; nn < 4; ++nn)
#pragma unroll
      for (int r = 0; r < 8; ++r) {
        const int t = m0 + r + 8 * side;
        dst[(size_t)(nn * 16 + col) * T_SEQ + t] = (__bf16)acc[nn][r];
      }
  }
}

// ---------------------------------------------------------------------------
// Kernel 2: flash attention. grid = (T/16, NH), block = 32 (one wave).
// S processed in 32-key tiles up to the causal diagonal. Logit cap + mask
// match the reference (tanh(l/50)*50, masked -> -100, softmax over all S).
// ---------------------------------------------------------------------------
__global__ __launch_bounds__(32) void attn_kernel(
    const __bf16* __restrict__ Qb, const __bf16* __restrict__ Kb,
    const __bf16* __restrict__ Vt, __bf16* __restrict__ Eb) {
  const int q0   = blockIdx.x * 16;
  const int head = blockIdx.y;
  const int kvh  = head >> 2;  // G = NH/NKV = 4
  const int lane = threadIdx.x & 31, col = lane & 15, side = lane >> 4;

  const __bf16* Qh = Qb + (size_t)head * T_SEQ * HD + (size_t)q0 * HD;
  const __bf16* Kh = Kb + (size_t)kvh * T_SEQ * HD;
  const __bf16* Vh = Vt + (size_t)kvh * HD * T_SEQ;  // [h][t]

  const v16bf aq0 = load_frag_row(Qh, HD);       // h = 0..31
  const v16bf aq1 = load_frag_row(Qh + 32, HD);  // h = 32..63

  v8f acc[4];
#pragma unroll
  for (int nn = 0; nn < 4; ++nn) acc[nn] = zero8();
  float mrow[8], drow[8];
#pragma unroll
  for (int r = 0; r < 8; ++r) { mrow[r] = -1e30f; drow[r] = 0.0f; }

  __shared__ __align__(16) __bf16 pS[16 * 32];

  const int s_last = q0 + 15;
  for (int s0 = 0; s0 <= s_last; s0 += 32) {
    // Prefetch next K/V tiles (lowers to global_prefetch_b8)
    if (s0 + 32 <= s_last) {
      __builtin_prefetch((const void*)(Kh + (size_t)(s0 + 32) * HD), 0, 1);
      __builtin_prefetch((const void*)(Vh + (size_t)(s0 + 32)), 0, 1);
    }
    // ---- logits: Q (16x64) @ K^T (64x32) -> 16x32 (K rows are contiguous)
    v8f lg[2];
    lg[0] = zero8();
    lg[1] = zero8();
#pragma unroll
    for (int nn = 0; nn < 2; ++nn) {
      v16bf b0 = load_frag_row(Kh + (size_t)(s0 + nn * 16) * HD, HD);
      lg[nn] = wmma_bf16(aq0, b0, lg[nn]);
      v16bf b1 = load_frag_row(Kh + (size_t)(s0 + nn * 16) * HD + 32, HD);
      lg[nn] = wmma_bf16(aq1, b1, lg[nn]);
    }
    // ---- cap + causal mask + online softmax stats
    float sc[8];
#pragma unroll
    for (int r = 0; r < 8; ++r) {
      const int t = q0 + r + 8 * side;
#pragma unroll
      for (int nn = 0; nn < 2; ++nn) {
        const int s = s0 + nn * 16 + col;
        const float capped = tanhf(lg[nn][r] * 0.02f) * 50.0f;
        lg[nn][r] = (s > t) ? -100.0f : capped;
      }
      float tm = fmaxf(lg[0][r], lg[1][r]);
#pragma unroll
      for (int off = 8; off > 0; off >>= 1) tm = fmaxf(tm, __shfl_xor(tm, off, 16));
      const float mnew = fmaxf(mrow[r], tm);
      const float esc  = __expf(mrow[r] - mnew);
      const float p0 = __expf(lg[0][r] - mnew);
      const float p1 = __expf(lg[1][r] - mnew);
      lg[0][r] = p0;
      lg[1][r] = p1;
      float rs = p0 + p1;
#pragma unroll
      for (int off = 8; off > 0; off >>= 1) rs += __shfl_xor(rs, off, 16);
      drow[r] = drow[r] * esc + rs;
      mrow[r] = mnew;
      sc[r] = esc;
    }
#pragma unroll
    for (int nn = 0; nn < 4; ++nn)
#pragma unroll
      for (int r = 0; r < 8; ++r) acc[nn][r] *= sc[r];

    // ---- C-layout probs -> A-layout fragment via LDS (b128 reload)
    __syncthreads();
#pragma unroll
    for (int nn = 0; nn < 2; ++nn)
#pragma unroll
      for (int r = 0; r < 8; ++r)
        pS[(r + 8 * side) * 32 + nn * 16 + col] = (__bf16)lg[nn][r];
    __syncthreads();
    const v16bf ap = load_frag_row(pS, 32);

    // ---- P (16x32) @ V (32x64); V^T rows contiguous -> 2x b128 per fragment
#pragma unroll
    for (int nn = 0; nn < 4; ++nn) {
      v16bf bv = load_frag_row(Vh + (size_t)(nn * 16) * T_SEQ + s0, T_SEQ);
      acc[nn] = wmma_bf16(ap, bv, acc[nn]);
    }
    __syncthreads();
  }

  // enc stored as T x DM bf16 matrix, column k = head*64 + h
#pragma unroll
  for (int nn = 0; nn < 4; ++nn)
#pragma unroll
    for (int r = 0; r < 8; ++r) {
      const int t = q0 + r + 8 * side;
      Eb[(size_t)t * DM + head * HD + nn * 16 + col] =
          (__bf16)(acc[nn][r] / drow[r]);
    }
}

// ---------------------------------------------------------------------------
// Kernel 3: output projection enc(2048x1024) @ wout(1024x1024) -> fp32 out.
// grid = (T/16, DM/64), block = 32 (one wave per 16x64 tile).
// ---------------------------------------------------------------------------
__global__ __launch_bounds__(32) void outproj_kernel(
    const __bf16* __restrict__ Eb, const __bf16* __restrict__ woutT,
    float* __restrict__ out) {
  const int m0 = blockIdx.x * 16;
  const int n0 = blockIdx.y * 64;
  v8f acc[4];
#pragma unroll
  for (int nn = 0; nn < 4; ++nn) acc[nn] = zero8();

  for (int kk = 0; kk < DM; kk += 32) {
    v16bf a = load_frag_row(Eb + (size_t)m0 * DM + kk, DM);
#pragma unroll
    for (int nn = 0; nn < 4; ++nn) {
      v16bf b = load_frag_row(woutT + (size_t)(n0 + nn * 16) * DM + kk, DM);
      acc[nn] = wmma_bf16(a, b, acc[nn]);
    }
  }
  const int lane = threadIdx.x & 31, col = lane & 15, side = lane >> 4;
#pragma unroll
  for (int nn = 0; nn < 4; ++nn)
#pragma unroll
    for (int r = 0; r < 8; ++r) {
      const int t = m0 + r + 8 * side;
      out[(size_t)t * DM + n0 + nn * 16 + col] = acc[nn][r];
    }
}

// ---------------------------------------------------------------------------
extern "C" void kernel_launch(void* const* d_in, const int* in_sizes, int n_in,
                              void* d_out, int out_size, void* d_ws, size_t ws_size,
                              hipStream_t stream) {
  (void)in_sizes; (void)n_in; (void)out_size; (void)ws_size;
  const float* x    = (const float*)d_in[0];
  const int*   pos  = (const int*)d_in[1];
  // d_in[2] = mask (strict upper-triangular causal) -- computed analytically
  const float* wq   = (const float*)d_in[3];
  const float* wkv  = (const float*)d_in[4];
  const float* wout = (const float*)d_in[5];
  float* out = (float*)d_out;

  // Workspace (bf16):
  //   Qb [16][2048][64] | Kb [4][2048][64] | Vt [4][64][2048] | Eb [2048][1024]
  //   xb [2048][1024]   | wqT [16][64][1024] | wkvT [8][64][1024] | woutT [1024][1024]
  __bf16* Qb    = (__bf16*)d_ws;
  __bf16* Kb    = Qb + (size_t)NH * T_SEQ * HD;
  __bf16* Vt    = Kb + (size_t)NKV * T_SEQ * HD;
  __bf16* Eb    = Vt + (size_t)NKV * HD * T_SEQ;
  __bf16* xb    = Eb + (size_t)T_SEQ * DM;
  __bf16* wqT   = xb + (size_t)T_SEQ * DM;
  __bf16* wkvT  = wqT + (size_t)NH * HD * DM;
  __bf16* woutT = wkvT + (size_t)2 * NKV * HD * DM;

  // --- pre-pack: x -> bf16; weights -> transposed bf16 [N][K]
  {
    const int n4 = T_SEQ * DM / 4;
    cvt_bf16_kernel<<<(n4 + 255) / 256, 256, 0, stream>>>(x, xb, n4);
  }
  {
    const int tot = NH * DM * HD;  // wq: 16 mats of (1024 x 64)
    transpose_cvt_kernel<<<(tot + 255) / 256, 256, 0, stream>>>(wq, wqT, DM, HD, tot);
  }
  {
    const int tot = 2 * NKV * DM * HD;  // wkv: 8 mats of (1024 x 64)
    transpose_cvt_kernel<<<(tot + 255) / 256, 256, 0, stream>>>(wkv, wkvT, DM, HD, tot);
  }
  {
    const int tot = DM * DM;  // wout: 1 mat of (1024 x 1024)
    transpose_cvt_kernel<<<(tot + 255) / 256, 256, 0, stream>>>(wout, woutT, DM, DM, tot);
  }

  qkv_rope_kernel<<<dim3(T_SEQ / 16, (NH + 2 * NKV) / 4), 128, 0, stream>>>(
      xb, pos, wqT, wkvT, Qb, Kb, Vt);
  attn_kernel<<<dim3(T_SEQ / 16, NH), 32, 0, stream>>>(Qb, Kb, Vt, Eb);
  outproj_kernel<<<dim3(T_SEQ / 16, DM / 64), 32, 0, stream>>>(Eb, woutT, out);
}
